// HRQuantizeEMAReset_12678743458264
// MI455X (gfx1250) — matrix operand order repaired and checked by reading, not compiled, not run on hardware
//
#include <hip/hip_runtime.h>
#include <hip/hip_bf16.h>
#include <math.h>

// ---------------------------------------------------------------------------
// HRQuantizeEMAReset forward for MI455X (gfx1250, wave32).
// Distance GEMM on v_wmma_f32_16x16x32_bf16, 32 rows/wave (2 WMMA per B load),
// manually software-pipelined B stream; everything else fp32.
// ---------------------------------------------------------------------------

typedef __bf16 bf16_t;
typedef __attribute__((ext_vector_type(8)))  __bf16 v8bf;
typedef __attribute__((ext_vector_type(16))) __bf16 v16bf;
typedef __attribute__((ext_vector_type(8)))  float  v8f;

#define NB       16
#define CDIM     512
#define TDIM     2048
#define KCODE    1024
#define NUMEL    (16777216ULL)       // 16*512*2048 = 2^24
#define TOTIDX   61440               // 32768+16384+8192+4096

// workspace layout (bytes), all offsets 256B aligned (~102 MB total)
#define OFF_RES   0ULL                       // fp32 residual  [16,512,2048]  64 MB
#define OFF_A     67108864ULL                // bf16 A matrix  [32768,512]    32 MB
#define OFF_CBB   100663296ULL               // bf16 codebook  [1024,512]      1 MB
#define OFF_CSQ   101711872ULL               // fp32 csq       [1024]
#define OFF_IDX   101715968ULL               // int idx        [61440]
#define OFF_CNT   101961728ULL               // fp32 counts    [1024]
#define OFF_LOSS  101965824ULL               // fp32 loss accumulator [1]

static __device__ __forceinline__ unsigned short f32_to_bf16_bits(float f) {
  unsigned int u = __float_as_uint(f);
  u = (u + 0x7FFFu + ((u >> 16) & 1u)) >> 16;  // round-to-nearest-even
  return (unsigned short)u;
}

// --- codebook: fp32 -> bf16 bits, plus per-code squared norm ---------------
__global__ __launch_bounds__(128) void k_cb(const float* __restrict__ cb,
                                            unsigned short* __restrict__ cbB,
                                            float* __restrict__ csq) {
  __shared__ float s[128];
  const int k = blockIdx.x;
  float sum = 0.f;
#pragma unroll
  for (int i = 0; i < 4; ++i) {
    int e = threadIdx.x + i * 128;
    float v = cb[(size_t)k * CDIM + e];
    cbB[(size_t)k * CDIM + e] = f32_to_bf16_bits(v);
    sum += v * v;
  }
  s[threadIdx.x] = sum;
  __syncthreads();
  for (int st = 64; st > 0; st >>= 1) {
    if (threadIdx.x < st) s[threadIdx.x] += s[threadIdx.x + st];
    __syncthreads();
  }
  if (threadIdx.x == 0) csq[k] = s[0];
}

__global__ void k_zero(float* counts, float* lossAccum) {
  counts[threadIdx.x] = 0.f;
  if (threadIdx.x == 0) lossAccum[0] = 0.f;
}

// --- avg-pool over T + transpose [N,C,T] -> bf16 A [N*t_s, C] --------------
// src = x for scale 1, residual otherwise.
__global__ __launch_bounds__(256) void k_down(const float* __restrict__ src,
                                              unsigned short* __restrict__ A,
                                              int scale, int t_s) {
  __shared__ float tile[32][33];
  const int n  = blockIdx.z;
  const int c0 = blockIdx.y * 32;
  const int t0 = blockIdx.x * 32;
  const int tq = threadIdx.x & 31;
  const int cq = threadIdx.x >> 5;   // 0..7
  const float inv = 1.0f / (float)scale;
#pragma unroll
  for (int i = 0; i < 4; ++i) {
    const int cl = cq + i * 8;
    const float* p = src + ((size_t)n * CDIM + (c0 + cl)) * TDIM
                         + (size_t)(t0 + tq) * scale;
    float s = 0.f;
    for (int j = 0; j < scale; ++j) s += p[j];
    tile[cl][tq] = s * inv;
  }
  __syncthreads();
#pragma unroll
  for (int i = 0; i < 4; ++i) {
    const int tl = cq + i * 8;
    A[(size_t)(n * t_s + t0 + tl) * CDIM + c0 + tq] = f32_to_bf16_bits(tile[tq][tl]);
  }
}

// --- WMMA distance-argmin: one wave per 32-row tile vs all 1024 codes ------
// Two A fragment sets share every B fragment (2 WMMAs per B load pair, halved
// codebook L2 traffic). B is manually double-buffered so the loads for the
// next k-tile (or the next code tile's first fragment) are in flight during
// the current WMMA pair.
__global__ __launch_bounds__(256) void k_argmin(const bf16_t* __restrict__ A,
                                                const bf16_t* __restrict__ cbB,
                                                const float* __restrict__ csq,
                                                int* __restrict__ idxOut, int M) {
  const int lane = threadIdx.x & 31;
  const int wave = threadIdx.x >> 5;
  const int rowBase = (blockIdx.x * 8 + wave) * 32;
  if (rowBase >= M) return;                 // wave-uniform: EXEC stays all-1s
  const int grp = lane >> 4;                // half-wave group
  const int l16 = lane & 15;

  // A 16x32 bf16 lane map: group0 holds K[0..15], group1 K[16..31]
  const bf16_t* arow0 = A + (size_t)(rowBase + l16) * CDIM;
  const bf16_t* arow1 = A + (size_t)(rowBase + 16 + l16) * CDIM;
  v16bf afrag0[16], afrag1[16];
#pragma unroll
  for (int kt = 0; kt < 16; ++kt) {
    const int kb = kt * 32 + grp * 16;
    v8bf p0 = *(const v8bf*)(arow0 + kb);
    v8bf p1 = *(const v8bf*)(arow0 + kb + 8);
    afrag0[kt] = __builtin_shufflevector(p0, p1,
                   0, 1, 2, 3, 4, 5, 6, 7, 8, 9, 10, 11, 12, 13, 14, 15);
    v8bf q0 = *(const v8bf*)(arow1 + kb);
    v8bf q1 = *(const v8bf*)(arow1 + kb + 8);
    afrag1[kt] = __builtin_shufflevector(q0, q1,
                   0, 1, 2, 3, 4, 5, 6, 7, 8, 9, 10, 11, 12, 13, 14, 15);
  }

  float minv0[8], minv1[8];
  int   mini0[8], mini1[8];
#pragma unroll
  for (int v = 0; v < 8; ++v) {
    minv0[v] = 3.4e38f; mini0[v] = 0;
    minv1[v] = 3.4e38f; mini1[v] = 0;
  }

  // B 32x16 bf16 lane map: lane holds column `l16`, K range = grp*16..grp*16+15
  v16bf bcur = *(const v16bf*)(cbB + (size_t)l16 * CDIM + grp * 16); // ct=0,kt=0

  for (int ct = 0; ct < 64; ++ct) {
    const int code = ct * 16 + l16;
    const bf16_t* brow = cbB + (size_t)code * CDIM + grp * 16;
    const int codeN = (ct < 63) ? (code + 16) : code;      // clamp keeps reads in bounds
    const bf16_t* brown = cbB + (size_t)codeN * CDIM + grp * 16;
    if (ct < 63) __builtin_prefetch((const void*)brown, 0, 1);

    v8f acc0 = {};
    v8f acc1 = {};
#pragma unroll
    for (int kt = 0; kt < 16; ++kt) {
      v16bf bnext;
      if (kt < 15) bnext = *(const v16bf*)(brow + (kt + 1) * 32);
      else         bnext = *(const v16bf*)(brown);          // next ct's kt=0
      acc0 = __builtin_amdgcn_wmma_f32_16x16x32_bf16(
                 false, afrag0[kt], false, bcur, (short)0, acc0, false, false);
      acc1 = __builtin_amdgcn_wmma_f32_16x16x32_bf16(
                 false, afrag1[kt], false, bcur, (short)0, acc1, false, false);
      bcur = bnext;
    }

    const float sc = csq[code];              // dist = csq[k] - 2*dot (+||row||², const per row)
#pragma unroll
    for (int v = 0; v < 8; ++v) {
      float s0 = fmaf(-2.0f, acc0[v], sc);
      if (s0 < minv0[v]) { minv0[v] = s0; mini0[v] = code; }
      float s1 = fmaf(-2.0f, acc1[v], sc);
      if (s1 < minv1[v]) { minv1[v] = s1; mini1[v] = code; }
    }
  }

  // C layout: lane group0 VGPR v -> row v, group1 -> row 8+v; columns = l16.
  // Min-reduce over the 16 lanes of each half-wave (xor masks stay in-group).
#pragma unroll
  for (int off = 1; off < 16; off <<= 1) {
#pragma unroll
    for (int v = 0; v < 8; ++v) {
      float ov = __shfl_xor(minv0[v], off, 32);
      int   oi = __shfl_xor(mini0[v], off, 32);
      if (ov < minv0[v] || (ov == minv0[v] && oi < mini0[v])) { minv0[v] = ov; mini0[v] = oi; }
      ov = __shfl_xor(minv1[v], off, 32);
      oi = __shfl_xor(mini1[v], off, 32);
      if (ov < minv1[v] || (ov == minv1[v] && oi < mini1[v])) { minv1[v] = ov; mini1[v] = oi; }
    }
  }
  if (l16 == 0) {
#pragma unroll
    for (int v = 0; v < 8; ++v) {
      idxOut[rowBase + grp * 8 + v]      = mini0[v];
      idxOut[rowBase + 16 + grp * 8 + v] = mini1[v];
    }
  }
}

// --- gather + linear upsample + residual/f_hat/loss update -----------------
// first=1 (scale 1): res_in = x, fhat_prev = 0 (arrays get their first write).
__global__ __launch_bounds__(256) void k_update(const float* __restrict__ x,
                                                const float* __restrict__ cb,
                                                const int* __restrict__ idx,
                                                float* __restrict__ res,
                                                float* __restrict__ fhat,
                                                float* __restrict__ lossAccum,
                                                int scale, int t_s, int first) {
  __shared__ float sdata[256];
  const size_t g = (size_t)blockIdx.x * 256 + threadIdx.x;
  const int t = (int)(g & (TDIM - 1));
  const int c = (int)((g >> 11) & (CDIM - 1));
  const int n = (int)(g >> 20);

  float pos = (t + 0.5f) / (float)scale - 0.5f;          // align_corners=False
  pos = fminf(fmaxf(pos, 0.f), (float)(t_s - 1));
  const int lo = (int)floorf(pos);
  const int hi = min(lo + 1, t_s - 1);
  const float w = pos - (float)lo;
  const int i0 = idx[n * t_s + lo];
  const int i1 = idx[n * t_s + hi];
  const float up = cb[(size_t)i0 * CDIM + c] * (1.f - w)
                 + cb[(size_t)i1 * CDIM + c] * w;

  const float xv = x[g];
  const float rin = first ? xv : res[g];
  res[g] = rin - up;
  const float f = (first ? 0.f : fhat[g]) + up;
  fhat[g] = f;
  const float d = xv - f;

  sdata[threadIdx.x] = d * d;
  __syncthreads();
  for (int s2 = 128; s2 > 0; s2 >>= 1) {
    if (threadIdx.x < s2) sdata[threadIdx.x] += sdata[threadIdx.x + s2];
    __syncthreads();
  }
  if (threadIdx.x == 0) atomicAdd(lossAccum, sdata[0]);
}

// --- histogram + perplexity/loss scalars -----------------------------------
__global__ void k_hist(const int* __restrict__ idx, float* __restrict__ counts,
                       int total) {
  int i = blockIdx.x * 256 + threadIdx.x;
  if (i < total) atomicAdd(&counts[idx[i]], 1.0f);
}

__global__ __launch_bounds__(1024) void k_final(const float* __restrict__ counts,
                                                const float* __restrict__ lossAccum,
                                                float* __restrict__ scalars) {
  __shared__ float s[1024];
  const float p = counts[threadIdx.x] / (float)TOTIDX;
  s[threadIdx.x] = p * logf(p + 1e-7f);
  __syncthreads();
  for (int st = 512; st > 0; st >>= 1) {
    if (threadIdx.x < st) s[threadIdx.x] += s[threadIdx.x + st];
    __syncthreads();
  }
  if (threadIdx.x == 0) {
    scalars[0] = lossAccum[0] / (4.0f * (float)NUMEL);   // mean_vq_loss
    scalars[1] = expf(-s[0]);                            // perplexity
  }
}

extern "C" void kernel_launch(void* const* d_in, const int* in_sizes, int n_in,
                              void* d_out, int out_size, void* d_ws, size_t ws_size,
                              hipStream_t stream) {
  const float* x  = (const float*)d_in[0];   // [16,512,2048]
  const float* cb = (const float*)d_in[1];   // [1024,512]
  float* out = (float*)d_out;                // f_hat (16M floats) + 2 scalars

  char* w = (char*)d_ws;                     // needs ~102 MB of workspace
  float*          res    = (float*)(w + OFF_RES);
  unsigned short* Abits  = (unsigned short*)(w + OFF_A);
  unsigned short* cbBits = (unsigned short*)(w + OFF_CBB);
  float*          csq    = (float*)(w + OFF_CSQ);
  int*            idxBuf = (int*)(w + OFF_IDX);
  float*          counts = (float*)(w + OFF_CNT);
  float*          loss   = (float*)(w + OFF_LOSS);

  k_cb<<<KCODE, 128, 0, stream>>>(cb, cbBits, csq);
  k_zero<<<1, 1024, 0, stream>>>(counts, loss);

  const int scales[4] = {1, 2, 4, 8};
  int idxOff = 0;
  for (int si = 0; si < 4; ++si) {
    const int sc  = scales[si];
    const int t_s = TDIM / sc;
    const int M   = NB * t_s;
    const float* src = (si == 0) ? x : (const float*)res;
    k_down<<<dim3(t_s / 32, CDIM / 32, NB), 256, 0, stream>>>(src, Abits, sc, t_s);
    k_argmin<<<M / 256, 256, 0, stream>>>((const bf16_t*)Abits,
                                          (const bf16_t*)cbBits, csq,
                                          idxBuf + idxOff, M);
    k_update<<<(unsigned)(NUMEL / 256), 256, 0, stream>>>(x, cb, idxBuf + idxOff,
                                                          res, out, loss, sc, t_s,
                                                          si == 0 ? 1 : 0);
    idxOff += M;
  }

  k_hist<<<(TOTIDX + 255) / 256, 256, 0, stream>>>(idxBuf, counts, TOTIDX);
  k_final<<<1, 1024, 0, stream>>>(counts, loss, out + NUMEL);
}
